// MultiHeadAttentionPySDPALlama2_22247930594001
// MI455X (gfx1250) — matrix operand order repaired and checked by reading, not compiled, not run on hardware
//
#include <hip/hip_runtime.h>
#include <hip/hip_bf16.h>

// MI455X (gfx1250) implementation of a Llama-2 attention block:
//   qkv = x @ w_qkv^T ; RoPE(q,k) ; causal flash-SDPA ; out = ctx @ w_proj^T
// All matmuls run on v_wmma_f32_16x16x32_f16 (f16 in, fp32 accumulate).

typedef _Float16 h8   __attribute__((ext_vector_type(8)));
typedef _Float16 v16h __attribute__((ext_vector_type(16)));
typedef float    v8f  __attribute__((ext_vector_type(8)));

#define NEG_BIG (-3.0e38f)

static __device__ __forceinline__ v8f wmma_16x16x32(v16h a, v16h b, v8f c) {
  // 8 args: (neg_a, A, neg_b, B, c_mod, C, reuse_a, reuse_b)
  return __builtin_amdgcn_wmma_f32_16x16x32_f16(false, a, false, b, (short)0, c,
                                                false, false);
}

// A-fragment (16x32, f16), ISA layout: lane holds row = rowBase + (lane&15);
// K halves: [k0..k0+7] and [k0+16..k0+23], k0 = kBase + 8*(lane>=16).
static __device__ __forceinline__ v16h
load_a_frag(const _Float16* __restrict__ base, int row, int ld, int kBase, int lane) {
  const int k0 = kBase + ((lane & 16) ? 8 : 0);
  const _Float16* p = base + (size_t)row * (size_t)ld + k0;
  h8 lo = *(const h8*)(p);
  h8 hi = *(const h8*)(p + 16);
  return __builtin_shufflevector(lo, hi, 0, 1, 2, 3, 4, 5, 6, 7,
                                 8, 9, 10, 11, 12, 13, 14, 15);
}

// B-fragment (32x16, f16): lane holds column n = colBase + (lane&15);
// K halves: [k0..k0+7] and [k0+16..k0+23], k0 = kBase + 8*(lane>=16). Source stored as (N x K).
static __device__ __forceinline__ v16h
load_b_frag(const _Float16* __restrict__ base, int col, int ld, int kBase, int lane) {
  const int k0 = kBase + ((lane & 16) ? 8 : 0);
  const _Float16* p = base + (size_t)col * (size_t)ld + k0;
  h8 lo = *(const h8*)(p);
  h8 hi = *(const h8*)(p + 16);
  return __builtin_shufflevector(lo, hi, 0, 1, 2, 3, 4, 5, 6, 7,
                                 8, 9, 10, 11, 12, 13, 14, 15);
}

// ---------------------------------------------------------------- converts --
__global__ void __launch_bounds__(256)
f32_to_f16_kernel(const float* __restrict__ in, _Float16* __restrict__ out, int n) {
  int i = blockIdx.x * 256 + threadIdx.x;
  if (i < n) out[i] = (_Float16)in[i];
}

// ------------------------------------------------------------------- GEMM ---
// C(MxN) = A(MxK) * Bt(NxK)^T.  One wave computes a 32x128 C tile:
// 16 WMMAs per 32-wide K chunk against 10 fragment loads (A amortized x8).
// blockDim = (32, 8). Grids divide exactly -> EXEC is all ones at every WMMA.
template <bool HALF_OUT>
__global__ void __launch_bounds__(256)
gemm_nt_wmma(const _Float16* __restrict__ A, const _Float16* __restrict__ Bt,
             void* __restrict__ Cout, int M, int N, int K) {
  const int lane = threadIdx.x;
  const long wid = (long)blockIdx.x * 8 + threadIdx.y;
  const int nTiles = N >> 7;                 // 128-wide N tiles
  const int m0 = (int)(wid / nTiles) << 5;   // 32-row M tile
  const int n0 = (int)(wid % nTiles) << 7;
  const int r16 = lane & 15, hi = lane >> 4;

  v8f acc[2][8];
#pragma unroll
  for (int mt = 0; mt < 2; ++mt)
#pragma unroll
    for (int nt = 0; nt < 8; ++nt) acc[mt][nt] = (v8f)0.0f;

  for (int kc = 0; kc < K; kc += 32) {
    // pull next K chunk toward the WGP (global_prefetch_b8; speculative,
    // OOB translations silently dropped per ISA)
    __builtin_prefetch(A + (size_t)(m0 + r16) * K + kc + 32, 0, 3);
    __builtin_prefetch(Bt + (size_t)(n0 + lane * 4) * K + kc + 32, 0, 3);

    v16h a0 = load_a_frag(A, m0 + r16, K, kc, lane);
    v16h a1 = load_a_frag(A, m0 + 16 + r16, K, kc, lane);
#pragma unroll
    for (int nt = 0; nt < 8; ++nt) {
      v16h b = load_b_frag(Bt, n0 + nt * 16 + r16, K, kc, lane);
      acc[0][nt] = wmma_16x16x32(a0, b, acc[0][nt]);
      acc[1][nt] = wmma_16x16x32(a1, b, acc[1][nt]);
    }
  }

#pragma unroll
  for (int mt = 0; mt < 2; ++mt)
#pragma unroll
    for (int nt = 0; nt < 8; ++nt)
#pragma unroll
      for (int r = 0; r < 8; ++r) {
        const size_t row = (size_t)(m0 + mt * 16 + r + 8 * hi);
        const size_t col = (size_t)(n0 + nt * 16 + r16);
        if constexpr (HALF_OUT)
          ((_Float16*)Cout)[row * N + col] = (_Float16)acc[mt][nt][r];
        else
          ((float*)Cout)[row * N + col] = acc[mt][nt][r];
      }
}

// -------------------------------------------------------- RoPE + reshape ----
// qkv: [B*S, 6144] f16 -> q,k: [B,H,S,128] f16 (rotated), vT: [B,H,128,S] f16.
__global__ void __launch_bounds__(256)
rope_reshape_kernel(const _Float16* __restrict__ qkv, _Float16* __restrict__ q,
                    _Float16* __restrict__ k, _Float16* __restrict__ vT) {
  const unsigned idx = blockIdx.x * 256 + threadIdx.x;  // ((b*16+h)*2048+s)*128+d
  const int d = idx & 127;
  const int s = (idx >> 7) & 2047;
  const int h = (idx >> 18) & 15;
  const int b = idx >> 22;
  const _Float16* row = qkv + (size_t)(b * 2048 + s) * 6144 + h * 128;
  const float qv = (float)row[d];
  const float kv = (float)row[2048 + d];
  const float vv = (float)row[4096 + d];
  const int dp = (d < 64) ? d + 64 : d - 64;
  const float sgn = (d < 64) ? -1.0f : 1.0f;
  const float qp = (float)row[dp];
  const float kp = (float)row[2048 + dp];
  const int j = d & 63;
  const float inv_freq = expf((float)j * (-0.14391156831212787f)); // -ln(1e4)/64
  const float ang = (float)s * inv_freq;
  const float c = cosf(ang), sn = sinf(ang);
  q[idx] = (_Float16)(qv * c + sgn * qp * sn);
  k[idx] = (_Float16)(kv * c + sgn * kp * sn);
  vT[(((size_t)(b * 16 + h) * 128 + d) << 11) + s] = (_Float16)vv;
}

// ------------------------------------------------------- flash attention ----
// One wave per (b, h, 16-row q tile). K-tiles of 32 columns, online softmax.
// blockDim = (32, 8); 8 waves/block, each with a private 16x40-half LDS tile.
__global__ void __launch_bounds__(256)
attn_kernel(const _Float16* __restrict__ q, const _Float16* __restrict__ k,
            const _Float16* __restrict__ vT, _Float16* __restrict__ ctx) {
  __shared__ _Float16 pbuf[8][16][40];  // stride 40 halves: 16B-aligned rows

  const int lane = threadIdx.x;
  const int wv = threadIdx.y;
  const int wid = blockIdx.x * 8 + wv;   // 0..4095
  const int qt = wid & 127;              // q tile (16 rows)
  const int h = (wid >> 7) & 15;
  const int b = wid >> 11;
  const int r16 = lane & 15, hi = lane >> 4;

  const _Float16* qb = q + (size_t)(b * 16 + h) * 2048 * 128;
  const _Float16* kb = k + (size_t)(b * 16 + h) * 2048 * 128;
  const _Float16* vb = vT + (size_t)(b * 16 + h) * 128 * 2048;

  v16h qf[4];
#pragma unroll
  for (int c = 0; c < 4; ++c) qf[c] = load_a_frag(qb, qt * 16 + r16, 128, c * 32, lane);

  v8f acc[8];
#pragma unroll
  for (int n = 0; n < 8; ++n) acc[n] = (v8f)0.0f;
  float m[8], l[8];
#pragma unroll
  for (int r = 0; r < 8; ++r) { m[r] = NEG_BIG; l[r] = 0.0f; }

  const float sc = 0.08838834764831845f;  // 1/sqrt(128)
  const int qmax = qt * 16 + 15;

  for (int kt = 0; kt * 32 <= qmax; ++kt) {
    // prefetch next k-tile of K and V^T while this tile computes
    __builtin_prefetch(kb + (size_t)(kt * 32 + 32 + lane) * 128, 0, 3);
    __builtin_prefetch(vb + (size_t)(lane * 4) * 2048 + kt * 32 + 32, 0, 3);

    // scores: Q(16x128) x K^T -> two 16x16 fp32 tiles (k cols 0..15 / 16..31)
    v8f st0 = (v8f)0.0f, st1 = (v8f)0.0f;
#pragma unroll
    for (int c = 0; c < 4; ++c) {
      v16h b0 = load_b_frag(kb, kt * 32 + r16, 128, c * 32, lane);
      st0 = wmma_16x16x32(qf[c], b0, st0);
    }
#pragma unroll
    for (int c = 0; c < 4; ++c) {
      v16h b1 = load_b_frag(kb, kt * 32 + 16 + r16, 128, c * 32, lane);
      st1 = wmma_16x16x32(qf[c], b1, st1);
    }

    const int kg0 = kt * 32 + r16, kg1 = kg0 + 16;
#pragma unroll
    for (int r = 0; r < 8; ++r) {
      const int qg = qt * 16 + r + 8 * hi;
      float s0 = (kg0 > qg) ? NEG_BIG : st0[r] * sc;   // causal mask (select)
      float s1 = (kg1 > qg) ? NEG_BIG : st1[r] * sc;
      // row max across the 16 lanes of this half (xor 1/2/4/8 keeps bit4)
      float t = fmaxf(s0, s1);
      t = fmaxf(t, __shfl_xor(t, 1, 32));
      t = fmaxf(t, __shfl_xor(t, 2, 32));
      t = fmaxf(t, __shfl_xor(t, 4, 32));
      t = fmaxf(t, __shfl_xor(t, 8, 32));
      const float mn = fmaxf(m[r], t);
      const float scale = __expf(m[r] - mn);
      const float p0 = __expf(s0 - mn);
      const float p1 = __expf(s1 - mn);
      float rs = p0 + p1;
      rs += __shfl_xor(rs, 1, 32);
      rs += __shfl_xor(rs, 2, 32);
      rs += __shfl_xor(rs, 4, 32);
      rs += __shfl_xor(rs, 8, 32);
      l[r] = l[r] * scale + rs;
      m[r] = mn;
#pragma unroll
      for (int n = 0; n < 8; ++n) acc[n][r] *= scale;   // rescale ctx rows
      const int row = r + 8 * hi;                       // C-layout row
      pbuf[wv][row][r16] = (_Float16)p0;
      pbuf[wv][row][r16 + 16] = (_Float16)p1;
    }
    // intra-wave LDS exchange: C-layout -> A-layout for P
    asm volatile("s_wait_dscnt 0x0" ::: "memory");
    v16h pf = load_a_frag(&pbuf[wv][0][0], r16, 40, 0, lane);

    // ctx += P(16x32) x V(32x128); V^T rows are contiguous over k
#pragma unroll
    for (int n = 0; n < 8; ++n) {
      v16h vf = load_b_frag(vb, n * 16 + r16, 2048, kt * 32, lane);
      acc[n] = wmma_16x16x32(pf, vf, acc[n]);
    }
  }

  // epilogue: normalize and write ctx as f16 [B, S, H*128]
#pragma unroll
  for (int n = 0; n < 8; ++n)
#pragma unroll
    for (int r = 0; r < 8; ++r) {
      const int qg = qt * 16 + r + 8 * hi;
      const float v = acc[n][r] / l[r];
      ctx[((size_t)(b * 2048 + qg)) * 2048 + h * 128 + n * 16 + r16] = (_Float16)v;
    }
}

// ---------------------------------------------------------------- launch ----
extern "C" void kernel_launch(void* const* d_in, const int* in_sizes, int n_in,
                              void* d_out, int out_size, void* d_ws, size_t ws_size,
                              hipStream_t stream) {
  (void)in_sizes; (void)n_in; (void)out_size; (void)ws_size;
  const float* x = (const float*)d_in[0];       // [2, 2048, 2048]
  const float* w_qkv = (const float*)d_in[1];   // [6144, 2048]
  const float* w_proj = (const float*)d_in[2];  // [2048, 2048]
  float* out = (float*)d_out;                   // [2, 2048, 2048]

  char* ws = (char*)d_ws;
  _Float16* xh   = (_Float16*)(ws + 0);                    // 16 MB
  _Float16* wqh  = (_Float16*)(ws + 16777216);             // 24 MB
  _Float16* wph  = (_Float16*)(ws + 41943040);             // 8 MB
  _Float16* qkvh = (_Float16*)(ws + 50331648);             // 48 MB
  _Float16* qh   = (_Float16*)(ws + 100663296);            // 16 MB
  _Float16* kh   = (_Float16*)(ws + 117440512);            // 16 MB
  _Float16* vth  = (_Float16*)(ws + 134217728);            // 16 MB
  _Float16* ctxh = (_Float16*)(ws + 150994944);            // 16 MB

  const int nX = 2 * 2048 * 2048;       // 8388608
  const int nWq = 6144 * 2048;          // 12582912
  const int nWp = 2048 * 2048;          // 4194304

  f32_to_f16_kernel<<<(nX + 255) / 256, 256, 0, stream>>>(x, xh, nX);
  f32_to_f16_kernel<<<(nWq + 255) / 256, 256, 0, stream>>>(w_qkv, wqh, nWq);
  f32_to_f16_kernel<<<(nWp + 255) / 256, 256, 0, stream>>>(w_proj, wph, nWp);

  // QKV GEMM: M=4096, N=6144, K=2048 -> (4096/32)*(6144/128)=6144 waves
  gemm_nt_wmma<true><<<6144 / 8, dim3(32, 8), 0, stream>>>(
      xh, wqh, (void*)qkvh, 4096, 6144, 2048);

  rope_reshape_kernel<<<nX / 256, 256, 0, stream>>>(qkvh, qh, kh, vth);

  // Flash attention: 2*16*128 = 4096 waves
  attn_kernel<<<4096 / 8, dim3(32, 8), 0, stream>>>(qh, kh, vth, ctxh);

  // Output projection: M=4096, N=2048, K=2048 -> (4096/32)*(2048/128)=2048 waves
  gemm_nt_wmma<false><<<2048 / 8, dim3(32, 8), 0, stream>>>(
      ctxh, wph, (void*)out, 4096, 2048, 2048);
}
